// PaRINet_16587163697675
// MI455X (gfx1250) — compile-verified, matrix operand order checked by reading, not run on hardware
//
#include <hip/hip_runtime.h>
#include <hip/hip_bf16.h>

// ---------------------------------------------------------------------------
// PaRI-Net forward for MI455X (gfx1250, wave32, WMMA).
// All large GEMMs use v_wmma_f32_16x16x32_f16 (f16 in, f32 acc).
// X tiles staged to LDS via the Tensor Data Mover (tensor_load_to_lds);
// fragments are built from f16 fragment-major LDS tiles with ds_load_b128.
// ---------------------------------------------------------------------------

#define Bb 8
#define Nn 2048
#define Kk 20
#define NKe (Nn * Kk)          // 40960 edge columns per batch
#define EPSB 1e-5f

typedef __attribute__((ext_vector_type(16))) _Float16 v16h;
typedef __attribute__((ext_vector_type(8)))  _Float16 v8h;
typedef __attribute__((ext_vector_type(8)))  float    v8f;
typedef __attribute__((ext_vector_type(4)))  unsigned int v4u;
typedef __attribute__((ext_vector_type(8)))  int      v8i;
typedef __attribute__((ext_vector_type(4)))  int      v4i;

#define CAT16(a, b) __builtin_shufflevector(a, b, 0, 1, 2, 3, 4, 5, 6, 7, \
                                            8, 9, 10, 11, 12, 13, 14, 15)

static constexpr int GM_STORE = 1;
static constexpr int GM_STATS = 2;
static constexpr int GM_MAXK  = 4;

// ------------------------------ helpers -----------------------------------

__device__ inline void atomicMaxF(float* addr, float val) {
  int* ia = (int*)addr;
  int cur = __float_as_int(*addr);
  while (__int_as_float(cur) < val) {
    int assumed = cur;
    cur = atomicCAS(ia, assumed, __float_as_int(val));
    if (cur == assumed) break;
  }
}

// TDM: DMA a 2D fp32 tile [rows x 128] (row stride gstride elems) into LDS.
// Descriptor layout per CDNA5 ISA ch.8 (group0 128b, group1 256b, 2D mode).
__device__ static inline void tdm_load_2d(void* ldsDst, const void* gsrc,
                                          int rows, long long gstrideElems) {
  unsigned lds = (unsigned)(unsigned long long)(size_t)ldsDst; // LDS offset = low 32 bits
  unsigned long long ga = (unsigned long long)(size_t)gsrc;
  v4u g0;
  g0[0] = 1u;                                            // count=1, user mode
  g0[1] = lds;                                           // lds_addr
  g0[2] = (unsigned)ga;                                  // global_addr[31:0]
  g0[3] = (unsigned)((ga >> 32) & 0x1FFFFFFu) | (2u << 30); // addr[56:32] | type=2
  unsigned long long s0 = (unsigned long long)gstrideElems;  // in data_size units
  unsigned td0 = 0x40000000u;                            // huge: tiles always in-bounds
  unsigned td1 = (unsigned)rows;
  v8i g1;
  g1[0] = (int)(2u << 16);                               // data_size = 4B
  g1[1] = (int)((td0 & 0xFFFFu) << 16);                  // tensor_dim0[15:0]
  g1[2] = (int)((td0 >> 16) | ((td1 & 0xFFFFu) << 16));  // dim0 hi | dim1 lo
  g1[3] = (int)((td1 >> 16) | (128u << 16));             // dim1 hi | tile_dim0=128
  g1[4] = (int)(td1 & 0xFFFFu);                          // tile_dim1=rows, tile_dim2=0
  g1[5] = (int)(unsigned)(s0 & 0xFFFFFFFFull);           // dim0_stride[31:0]
  g1[6] = (int)(unsigned)((s0 >> 32) & 0xFFFFull);       // dim0_stride[47:32]
  g1[7] = 0;
  v4i z4 = {0, 0, 0, 0};
#if defined(__clang_major__) && (__clang_major__ >= 23)
  v8i z8 = {0, 0, 0, 0, 0, 0, 0, 0};
  __builtin_amdgcn_tensor_load_to_lds(g0, g1, z4, z4, z8, 0);
#else
  __builtin_amdgcn_tensor_load_to_lds(g0, g1, z4, z4, 0);
#endif
}

// ----------------------------- WMMA GEMM -----------------------------------
// Y[b,o,m] = act(BN(sum_c W[o,c] * X'[b,c,m] + bias[o] + q[b,o,m/Kgrp]))
// X'[c,m] = optional input-BN+relu of X (feeds post-BN activations without an
// extra elementwise pass). MODE bits: stats accumulate / store / max-over-k.
template <int MODE, bool UTDM>
__global__ __launch_bounds__(256)
void k_gemm(const float* __restrict__ W, int ldw, const float* __restrict__ bias,
            const float* __restrict__ X, long long xbStride,
            const float* __restrict__ inM, const float* __restrict__ inI,
            const float* __restrict__ inG, const float* __restrict__ inB,
            const float* __restrict__ q, int qO,
            float* __restrict__ Y, int oStride,
            const float* __restrict__ bnM, const float* __restrict__ bnI,
            const float* __restrict__ bnG, const float* __restrict__ bnB, int act,
            float* __restrict__ stS, float* __restrict__ stQ,
            int C, int O, int M, int Kgrp) {
  __shared__ _Float16 sW[16][32];     // weights tile, fragment-major (row, k)
  __shared__ float    sXf[32][128];   // fp32 X tile as DMA'd / loaded (k, col)
  __shared__ _Float16 sXh[128][32];   // f16 X tile, fragment-major (col, k)
  __shared__ float    sE[16][128];
  __shared__ float    rS[16], rQ[16];

  const int tid = threadIdx.x;
  const int lane = tid & 31;
  const int wv = tid >> 5;
  const int b = blockIdx.z;
  const int o0 = blockIdx.y * 16;
  const int m0 = blockIdx.x * 128;
  const float* Xb = X + (long long)b * xbStride;

  __builtin_prefetch(&W[(size_t)o0 * ldw], 0, 3);

  const int lr = lane & 15;
  const int kb = (lane >> 4) << 3;    // {0, 8}
  const int colL = (wv << 4) + lr;    // this lane's column within the 128-tile
  v8f acc = {};
  const int nch = (C + 31) >> 5;
  for (int cc = 0; cc < nch; ++cc) {
    const int c0 = cc << 5;
    __syncthreads();
    // weights tile (f32 -> f16), fragment-major
    for (int t = tid; t < 16 * 32; t += 256) {
      int r = t >> 5, c = t & 31;
      float wval = 0.f;
      if (c0 + c < C) wval = W[(size_t)(o0 + r) * ldw + c0 + c];
      sW[r][c] = (_Float16)wval;
    }
    int rows = C - c0; if (rows > 32) rows = 32;
    if (UTDM) {
      if (wv == 0) {
        tdm_load_2d(&sXf[0][0], Xb + (size_t)c0 * M + m0, rows, (long long)M);
        __builtin_amdgcn_s_wait_tensorcnt(0);
      }
      for (int t = tid; t < (32 - rows) * 128; t += 256)
        sXf[rows + (t >> 7)][t & 127] = 0.f;
    } else {
      for (int t = tid; t < 32 * 128; t += 256) {
        int r = t >> 7, m = t & 127;
        float xv = 0.f;
        if (r < rows) xv = Xb[(size_t)(c0 + r) * M + m0 + m];
        sXf[r][m] = xv;
      }
    }
    __syncthreads();
    // convert to f16 fragment-major; fold optional input BN+relu
    for (int t = tid; t < 32 * 128; t += 256) {
      int r = t >> 7, m = t & 127;
      float v = sXf[r][m];
      if (inM) {
        int c = c0 + r;
        if (r < rows) {
          v = (v - inM[c]) * inI[c] * inG[c] + inB[c];
          v = fmaxf(v, 0.f);
        } else {
          v = 0.f;
        }
      }
      sXh[m][r] = (_Float16)v;
    }
    __syncthreads();
    // fragments: two aligned ds_load_b128 each
    v8h a0 = *(const v8h*)&sW[lr][kb];
    v8h a1 = *(const v8h*)&sW[lr][kb + 16];
    v8h b0 = *(const v8h*)&sXh[colL][kb];
    v8h b1 = *(const v8h*)&sXh[colL][kb + 16];
    v16h af = CAT16(a0, a1);
    v16h bf = CAT16(b0, b1);
    acc = __builtin_amdgcn_wmma_f32_16x16x32_f16(false, af, false, bf,
                                                 (short)0, acc, false, false);
  }

  // epilogue: lane holds (o = o0 + v + 8*hi, m = m0 + colL)
  const int hi = lane >> 4;
  const int m = m0 + colL;
  float vals[8];
#pragma unroll
  for (int v = 0; v < 8; ++v) {
    int o = o0 + v + (hi << 3);
    float y = acc[v];
    if (o < O) {
      if (bias) y += bias[o];
      if (q) y += q[((size_t)b * qO + o) * (M / Kgrp) + (m / Kgrp)];
    }
    vals[v] = y;
  }
  if (MODE & GM_STATS) {
    __syncthreads();
    if (tid < 16) { rS[tid] = 0.f; rQ[tid] = 0.f; }
    __syncthreads();
#pragma unroll
    for (int v = 0; v < 8; ++v) {
      int r = v + (hi << 3);
      if (o0 + r < O && m < M) {
        atomicAdd(&rS[r], vals[v]);
        atomicAdd(&rQ[r], vals[v] * vals[v]);
      }
    }
    __syncthreads();
    if (tid < 16 && (o0 + tid) < O) {
      atomicAdd(&stS[o0 + tid], rS[tid]);
      atomicAdd(&stQ[o0 + tid], rQ[tid]);
    }
  }
  if (MODE & (GM_STORE | GM_MAXK)) {
#pragma unroll
    for (int v = 0; v < 8; ++v) {
      int o = o0 + v + (hi << 3);
      float y = vals[v];
      if (o < O) {
        if (bnM) y = (y - bnM[o]) * bnI[o] * bnG[o] + bnB[o];
        if (act == 1) y = fmaxf(y, 0.f);
        else if (act == 2) y = (y >= 0.f) ? y : 0.2f * y;
      }
      vals[v] = y;
    }
    if (MODE & GM_STORE) {
#pragma unroll
      for (int v = 0; v < 8; ++v) {
        int o = o0 + v + (hi << 3);
        if (o < O && m < M) Y[((size_t)b * oStride + o) * M + m] = vals[v];
      }
    }
    if (MODE & GM_MAXK) {
      __syncthreads();
#pragma unroll
      for (int v = 0; v < 8; ++v) {
        int r = v + (hi << 3);
        sE[r][colL] = (o0 + r < O && m < M) ? vals[v] : -1e30f;
      }
      __syncthreads();
      const int Ng = M / Kgrp;
      int gFirst = m0 / Kgrp;
      int gLast = (m0 + 127) / Kgrp;
      int ngrp = gLast - gFirst + 1;
      for (int p = tid; p < 16 * ngrp; p += 256) {
        int r = p & 15, gi = p >> 4;
        int g = gFirst + gi;
        if (g >= Ng) continue;
        int s = g * Kgrp; if (s < m0) s = m0;
        int e2 = (g + 1) * Kgrp; if (e2 > m0 + 128) e2 = m0 + 128;
        float mx = -1e30f;
        for (int mm = s; mm < e2; ++mm) mx = fmaxf(mx, sE[r][mm - m0]);
        int o = o0 + r;
        if (o < O) {
          float* dst = &Y[((size_t)b * oStride + o) * Ng + g];
          bool full = (g * Kgrp >= m0) && ((g + 1) * Kgrp <= m0 + 128);
          if (full) *dst = mx;
          else atomicMaxF(dst, mx);
        }
      }
    }
  }
}

// ------------------------------ kNN ----------------------------------------
// WMMA Gram tiles (LDS-staged, f16 fragment-major) + fused top-20 selection.
// Candidate lists live in LDS so dynamic-index updates are single ds ops
// (register arrays would lower to 20-way v_cmp/select cascades).
template <int NCH>
__global__ __launch_bounds__(128)
void k_knn(const float* __restrict__ F, long long fbStride, int C,
           const float* __restrict__ xx, int* __restrict__ idxOut) {
  constexpr int K32 = NCH * 32;
  __shared__ _Float16 sA[16][K32];
  __shared__ _Float16 sB[64][K32];
  __shared__ float sD[16][64];
  __shared__ float sXX[64];
  __shared__ float sBD[16][Kk];
  __shared__ int   sBI[16][Kk];
  const int tid = threadIdx.x, lane = tid & 31, wv = tid >> 5;
  const int b = blockIdx.y, r0 = blockIdx.x * 16;
  const float* Fb = F + (long long)b * fbStride;

  for (int t = tid; t < 16 * K32; t += 128) {
    int r = t / K32, k = t % K32;
    float v = (k < C) ? Fb[(size_t)k * Nn + r0 + r] : 0.f;
    sA[r][k] = (_Float16)v;
  }
  for (int t = tid; t < 16 * Kk; t += 128) {
    sBD[t / Kk][t % Kk] = -1e30f;
    sBI[t / Kk][t % Kk] = 0;
  }
  __syncthreads();

  const int lr = lane & 15, kb = (lane >> 4) << 3;
  v16h afr[NCH];
#pragma unroll
  for (int cc = 0; cc < NCH; ++cc) {
    v8h a0 = *(const v8h*)&sA[lr][cc * 32 + kb];
    v8h a1 = *(const v8h*)&sA[lr][cc * 32 + kb + 16];
    afr[cc] = CAT16(a0, a1);
  }
  float bmin = -1e30f; int barg = 0;
  float rxx = (tid < 16) ? xx[b * Nn + r0 + tid] : 0.f;

  for (int t0 = 0; t0 < Nn; t0 += 64) {
    __syncthreads();   // previous selection / fragment reads done
    for (int t = tid; t < 64 * K32; t += 128) {
      int cl = t / K32, k = t % K32;
      float v = (k < C) ? Fb[(size_t)k * Nn + t0 + cl] : 0.f;
      sB[cl][k] = (_Float16)v;
    }
    if (tid < 64) sXX[tid] = xx[b * Nn + t0 + tid];
    __syncthreads();
    v8f acc = {};
#pragma unroll
    for (int cc = 0; cc < NCH; ++cc) {
      v8h b0 = *(const v8h*)&sB[(wv << 4) + lr][cc * 32 + kb];
      v8h b1 = *(const v8h*)&sB[(wv << 4) + lr][cc * 32 + kb + 16];
      v16h bf = CAT16(b0, b1);
      acc = __builtin_amdgcn_wmma_f32_16x16x32_f16(false, afr[cc], false, bf,
                                                   (short)0, acc, false, false);
    }
    const int hi = lane >> 4;
#pragma unroll
    for (int v = 0; v < 8; ++v) sD[v + (hi << 3)][(wv << 4) + lr] = acc[v];
    __syncthreads();
    if (tid < 16) {
      int row = r0 + tid;
      for (int j = 0; j < 64; ++j) {
        int colg = t0 + j;
        if (colg == row) continue;               // drop self (top-1 in ref)
        float d = 2.f * sD[tid][j] - rxx - sXX[j];
        if (d > bmin) {
          sBD[tid][barg] = d; sBI[tid][barg] = colg;
          bmin = sBD[tid][0]; barg = 0;
          for (int i = 1; i < Kk; ++i) {
            float qv = sBD[tid][i];
            if (qv < bmin) { bmin = qv; barg = i; }
          }
        }
      }
    }
  }
  __syncthreads();
  if (tid < 16) {
    int row = r0 + tid;
    for (int i = 0; i < Kk; ++i)
      idxOut[((size_t)b * Nn + row) * Kk + i] = sBI[tid][i];
  }
  if (__builtin_amdgcn_cluster_id_x() != 0) __builtin_amdgcn_s_cluster_barrier();
}

// --------------------------- small kernels ----------------------------------

__global__ void k_fill(float* p, float v, size_t n) {
  size_t i = (size_t)blockIdx.x * 256 + threadIdx.x;
  if (i < n) p[i] = v;
}

__global__ void k_tr3(const float* __restrict__ in, float* __restrict__ out) {
  int i = blockIdx.x * 256 + threadIdx.x;
  if (i >= Bb * Nn * 3) return;
  int c = i % 3; int n = (i / 3) % Nn; int b = i / (3 * Nn);
  out[((size_t)b * 3 + c) * Nn + n] = in[i];
}

__global__ void k_xx(const float* __restrict__ F, long long fbStride, int C,
                     float* __restrict__ xx) {
  int i = blockIdx.x * 256 + threadIdx.x;
  if (i >= Bb * Nn) return;
  int b = i / Nn, n = i % Nn;
  const float* Fb = F + (long long)b * fbStride;
  float s = 0.f;
  for (int c = 0; c < C; ++c) { float v = Fb[(size_t)c * Nn + n]; s += v * v; }
  xx[i] = s;
}

struct F3 { float x, y, z; };
__device__ inline F3 f3sub(F3 a, F3 b) { return {a.x - b.x, a.y - b.y, a.z - b.z}; }
__device__ inline F3 f3cross(F3 a, F3 b) {
  return {a.y * b.z - a.z * b.y, a.z * b.x - a.x * b.z, a.x * b.y - a.y * b.x};
}
__device__ inline float f3dot(F3 a, F3 b) { return a.x * b.x + a.y * b.y + a.z * b.z; }
__device__ inline float f3norm(F3 a) { return sqrtf(f3dot(a, a)); }
__device__ inline F3 f3scale(F3 a, float s) { return {a.x * s, a.y * s, a.z * s}; }
__device__ inline F3 ld3(const float* p, int i) { return {p[3 * i], p[3 * i + 1], p[3 * i + 2]}; }
__device__ inline float lfa(F3 pi, F3 pj, F3 nrm, F3 axis) {
  F3 ps = f3sub(pj, pi);
  F3 inp = f3scale(f3cross(nrm, ps), 1.f / fmaxf(f3norm(ps), 1e-12f));
  F3 st = f3scale(f3cross(nrm, axis), 1.f / fmaxf(f3norm(axis), 1e-12f));
  F3 c = f3cross(st, inp);
  float s = f3norm(c);
  if (f3dot(c, nrm) > 0.f) s = -s;
  return atan2f(s, f3dot(st, inp));
}

__global__ void k_appf(const float* __restrict__ pos, const float* __restrict__ l0,
                       const float* __restrict__ l1, const int* __restrict__ idx,
                       float* __restrict__ A) {
  int i = blockIdx.x * 256 + threadIdx.x;
  if (i >= Bb * Nn * Kk) return;
  int k = i % Kk; int n = (i / Kk) % Nn; int b = i / (Nn * Kk);
  int j = idx[i];
  int ri = b * Nn + n, rj = b * Nn + j;
  F3 pi = ld3(pos, ri), pj = ld3(pos, rj);
  F3 ni = ld3(l0, ri), nj = ld3(l0, rj);
  F3 ai = ld3(l1, ri), aj = ld3(l1, rj);
  F3 ps = f3sub(pj, pi);
  float d1 = f3norm(ps);
  F3 c = f3cross(ni, ps); float g1 = atan2f(f3norm(c), f3dot(ni, ps));
  c = f3cross(nj, ps);    float g2 = atan2f(f3norm(c), f3dot(nj, ps));
  c = f3cross(ni, nj);
  float sg = (f3dot(c, ps) < 0.f) ? -1.f : 1.f;
  float g3 = atan2f(f3norm(c) * sg, f3dot(ni, nj));
  float aij = lfa(pi, pj, ni, ai);
  float aji = lfa(pj, pi, nj, aj);
  int m = n * Kk + k;
  size_t bo = (size_t)b * 8 * NKe;
  A[bo + 0 * (size_t)NKe + m] = cosf(aij);
  A[bo + 1 * (size_t)NKe + m] = sinf(aij);
  A[bo + 2 * (size_t)NKe + m] = cosf(aji);
  A[bo + 3 * (size_t)NKe + m] = sinf(aji);
  A[bo + 4 * (size_t)NKe + m] = d1;
  A[bo + 5 * (size_t)NKe + m] = cosf(g1);
  A[bo + 6 * (size_t)NKe + m] = cosf(g2);
  A[bo + 7 * (size_t)NKe + m] = cosf(g3);
}

__global__ void k_x14(const float* __restrict__ x, const float* __restrict__ A,
                      const int* __restrict__ idx, float* __restrict__ X14) {
  int i = blockIdx.x * 256 + threadIdx.x;
  if (i >= Bb * Nn * Kk) return;
  int k = i % Kk; int n = (i / Kk) % Nn; int b = i / (Nn * Kk);
  int m = n * Kk + k;
  int j = idx[i];
  int ri = b * Nn + n, rj = b * Nn + j;
  size_t bo = (size_t)b * 14 * NKe;
  for (int c = 0; c < 8; ++c)
    X14[bo + (size_t)c * NKe + m] = A[((size_t)b * 8 + c) * NKe + m];
  for (int c = 0; c < 3; ++c) {
    float xi = x[ri * 3 + c], xj = x[rj * 3 + c];
    X14[bo + (size_t)(8 + c) * NKe + m] = xj - xi;
    X14[bo + (size_t)(11 + c) * NKe + m] = xi;
  }
}

// feat_pre = dk * gather(basis, idx), in place; per-channel stats accumulated.
__global__ __launch_bounds__(256)
void k_gathermul(float* __restrict__ dk, const float* __restrict__ basis,
                 const int* __restrict__ idx, int cin,
                 float* __restrict__ stS, float* __restrict__ stQ) {
  __shared__ float ls[128], lq[128];
  int tid = threadIdx.x;
  for (int t = tid; t < cin; t += 256) { ls[t] = 0.f; lq[t] = 0.f; }
  __syncthreads();
  size_t i = (size_t)blockIdx.x * 256 + tid;
  size_t total = (size_t)Bb * cin * NKe;
  if (i < total) {
    int m = (int)(i % NKe);
    size_t t2 = i / NKe;
    int c = (int)(t2 % cin);
    int b = (int)(t2 / cin);
    int n = m / Kk, k = m % Kk;
    int j = idx[((size_t)b * Nn + n) * Kk + k];
    float v = dk[i] * basis[((size_t)b * cin + c) * Nn + j];
    dk[i] = v;
    atomicAdd(&ls[c], v);
    atomicAdd(&lq[c], v * v);
  }
  __syncthreads();
  for (int t = tid; t < cin; t += 256) {
    if (ls[t] != 0.f || lq[t] != 0.f) {
      atomicAdd(&stS[t], ls[t]);
      atomicAdd(&stQ[t], lq[t]);
    }
  }
}

__global__ void k_makewq(const float* __restrict__ ec, int cin, int cout,
                         float* __restrict__ wq) {
  int i = blockIdx.x * 256 + threadIdx.x;
  if (i >= cout * cin) return;
  int o = i / cin, c = i % cin;
  wq[i] = ec[(size_t)o * 2 * cin + cin + c] - ec[(size_t)o * 2 * cin + c];
}

__global__ void k_finstat(const float* __restrict__ s, const float* __restrict__ q,
                          float* __restrict__ mean, float* __restrict__ inv,
                          float count, int C) {
  int i = blockIdx.x * 256 + threadIdx.x;
  if (i < C) {
    float m = s[i] / count;
    mean[i] = m;
    float v = q[i] / count - m * m;
    inv[i] = rsqrtf(fmaxf(v, 0.f) + EPSB);
  }
}

__global__ __launch_bounds__(256)
void k_pool(const float* __restrict__ c5, const float* __restrict__ m,
            const float* __restrict__ inv, const float* __restrict__ g,
            const float* __restrict__ b, float* __restrict__ gv) {
  __shared__ float sM[256], sS[256];
  int o = blockIdx.x, bi = blockIdx.y, tid = threadIdx.x;
  const float* row = c5 + ((size_t)bi * 1024 + o) * Nn;
  float mn = m[o], iv = inv[o] * g[o], bb = b[o];
  float mx = -1e30f, sm = 0.f;
  for (int n = tid; n < Nn; n += 256) {
    float y = (row[n] - mn) * iv + bb;
    y = (y >= 0.f) ? y : 0.2f * y;
    mx = fmaxf(mx, y); sm += y;
  }
  sM[tid] = mx; sS[tid] = sm;
  __syncthreads();
  for (int s2 = 128; s2 > 0; s2 >>= 1) {
    if (tid < s2) { sM[tid] = fmaxf(sM[tid], sM[tid + s2]); sS[tid] += sS[tid + s2]; }
    __syncthreads();
  }
  if (tid == 0) {
    gv[bi * 2048 + o] = sM[0];
    gv[bi * 2048 + 1024 + o] = sS[0] * (1.f / Nn);
  }
}

// Final FC head: 2048->512 (BN6+lrelu) -> 256 (BN7+lrelu) -> 40. Tiny; single block.
__global__ __launch_bounds__(256)
void k_fc(const float* __restrict__ g, const float* __restrict__ w1,
          const float* __restrict__ bn6g, const float* __restrict__ bn6b,
          const float* __restrict__ w2, const float* __restrict__ b2,
          const float* __restrict__ bn7g, const float* __restrict__ bn7b,
          const float* __restrict__ w3, const float* __restrict__ b3,
          float* __restrict__ out) {
  __shared__ float h1[8 * 512];
  __shared__ float h2[8 * 256];
  const int tid = threadIdx.x;
  for (int i = tid; i < 8 * 512; i += 256) {
    int bi = i >> 9, o = i & 511;
    const float* gr = g + bi * 2048;
    const float* wr = w1 + (size_t)o * 2048;
    float s = 0.f;
    for (int c = 0; c < 2048; ++c) s += gr[c] * wr[c];
    h1[i] = s;
  }
  __syncthreads();
  for (int o = tid; o < 512; o += 256) {
    float m = 0.f;
    for (int bi = 0; bi < 8; ++bi) m += h1[bi * 512 + o];
    m *= 0.125f;
    float v = 0.f;
    for (int bi = 0; bi < 8; ++bi) { float d = h1[bi * 512 + o] - m; v += d * d; }
    v *= 0.125f;
    float iv = rsqrtf(v + EPSB) * bn6g[o];
    for (int bi = 0; bi < 8; ++bi) {
      float y = (h1[bi * 512 + o] - m) * iv + bn6b[o];
      h1[bi * 512 + o] = (y >= 0.f) ? y : 0.2f * y;
    }
  }
  __syncthreads();
  for (int i = tid; i < 8 * 256; i += 256) {
    int bi = i >> 8, o = i & 255;
    const float* wr = w2 + (size_t)o * 512;
    float s = b2[o];
    for (int c = 0; c < 512; ++c) s += h1[bi * 512 + c] * wr[c];
    h2[i] = s;
  }
  __syncthreads();
  for (int o = tid; o < 256; o += 256) {
    float m = 0.f;
    for (int bi = 0; bi < 8; ++bi) m += h2[bi * 256 + o];
    m *= 0.125f;
    float v = 0.f;
    for (int bi = 0; bi < 8; ++bi) { float d = h2[bi * 256 + o] - m; v += d * d; }
    v *= 0.125f;
    float iv = rsqrtf(v + EPSB) * bn7g[o];
    for (int bi = 0; bi < 8; ++bi) {
      float y = (h2[bi * 256 + o] - m) * iv + bn7b[o];
      h2[bi * 256 + o] = (y >= 0.f) ? y : 0.2f * y;
    }
  }
  __syncthreads();
  for (int i = tid; i < 8 * 40; i += 256) {
    int bi = i / 40, o = i % 40;
    const float* wr = w3 + (size_t)o * 256;
    float s = b3[o];
    for (int c = 0; c < 256; ++c) s += h2[bi * 256 + c] * wr[c];
    out[i] = s;
  }
}

// ------------------------------- host --------------------------------------

extern "C" void kernel_launch(void* const* d_in, const int* in_sizes, int n_in,
                              void* d_out, int out_size, void* d_ws, size_t ws_size,
                              hipStream_t stream) {
  (void)in_sizes; (void)n_in; (void)out_size; (void)ws_size;
  const float* x   = (const float*)d_in[0];
  const float* pos = (const float*)d_in[1];
  const float* l0  = (const float*)d_in[2];
  const float* l1  = (const float*)d_in[3];
  const float* conv1_w = (const float*)d_in[4];
  const float* bn1_g = (const float*)d_in[5];
  const float* bn1_b = (const float*)d_in[6];
  const float* conv5_w = (const float*)d_in[43];
  const float* bn5_g = (const float*)d_in[44];
  const float* bn5_b = (const float*)d_in[45];
  const float* l1_w = (const float*)d_in[46];
  const float* bn6_g = (const float*)d_in[47];
  const float* bn6_b = (const float*)d_in[48];
  const float* l2_w = (const float*)d_in[49];
  const float* l2_b = (const float*)d_in[50];
  const float* bn7_g = (const float*)d_in[51];
  const float* bn7_b = (const float*)d_in[52];
  const float* l3_w = (const float*)d_in[53];
  const float* l3_b = (const float*)d_in[54];

  float* ws = (float*)d_ws;
  size_t off = 0;
  auto take = [&](size_t n) { size_t o = off; off += (n + 255) & ~(size_t)255; return o; };
  float* xcat  = ws + take((size_t)Bb * 512 * Nn);   // x1|x2|x3|x4 channel slices
  float* posT  = ws + take((size_t)Bb * 3 * Nn);
  float* xxb   = ws + take((size_t)Bb * Nn);
  int*   idxb  = (int*)(ws + take((size_t)Bb * Nn * Kk));
  float* Ab    = ws + take((size_t)Bb * 8 * NKe);
  float* X14   = ws + take((size_t)Bb * 14 * NKe);
  float* hbuf  = ws + take((size_t)Bb * 64 * NKe);
  float* dkbuf = ws + take((size_t)Bb * 128 * NKe);
  float* basb  = ws + take((size_t)Bb * 128 * Nn);
  float* qbuf  = ws + take((size_t)Bb * 256 * Nn);
  float* wqb   = ws + take((size_t)256 * 128);
  float* c5b   = ws + take((size_t)Bb * 1024 * Nn);
  float* gvec  = ws + take((size_t)Bb * 2048);
  float* stA   = ws + take(4096);
  float* stB   = ws + take(4096);
  float* stC   = ws + take(4096);
  float* stP   = ws + take(4096);

  const int EDGE = Bb * Nn * Kk;
  auto fill = [&](float* p, float v, size_t n) {
    k_fill<<<dim3((unsigned)((n + 255) / 256)), dim3(256), 0, stream>>>(p, v, n);
  };
  auto knn = [&](const float* F, long long fbs, int C) {
    k_xx<<<dim3((Bb * Nn + 255) / 256), 256, 0, stream>>>(F, fbs, C, xxb);
    if (C <= 32)
      k_knn<1><<<dim3(Nn / 16, Bb), 128, 0, stream>>>(F, fbs, C, xxb, idxb);
    else if (C <= 64)
      k_knn<2><<<dim3(Nn / 16, Bb), 128, 0, stream>>>(F, fbs, C, xxb, idxb);
    else
      k_knn<4><<<dim3(Nn / 16, Bb), 128, 0, stream>>>(F, fbs, C, xxb, idxb);
  };

  // -------- conv1 block --------
  fill(xcat, -1e30f, (size_t)Bb * 512 * Nn);  // max-k boundary atomics need init
  k_tr3<<<dim3((Bb * Nn * 3 + 255) / 256), 256, 0, stream>>>(pos, posT);
  knn(posT, 3ll * Nn, 3);
  k_appf<<<dim3((EDGE + 255) / 256), 256, 0, stream>>>(pos, l0, l1, idxb, Ab);
  k_x14<<<dim3((EDGE + 255) / 256), 256, 0, stream>>>(x, Ab, idxb, X14);
  fill(stA, 0.f, 2048);
  k_gemm<GM_STATS, false><<<dim3(NKe / 128, 4, Bb), 256, 0, stream>>>(
      conv1_w, 14, nullptr, X14, 14ll * NKe,
      nullptr, nullptr, nullptr, nullptr, nullptr, 0, nullptr, 0,
      nullptr, nullptr, nullptr, nullptr, 0, stA, stA + 1024, 14, 64, NKe, Kk);
  k_finstat<<<dim3(4), 256, 0, stream>>>(stA, stA + 1024, stA + 2048, stA + 3072,
                                         (float)(Bb * NKe), 64);
  k_gemm<GM_MAXK, false><<<dim3(NKe / 128, 4, Bb), 256, 0, stream>>>(
      conv1_w, 14, nullptr, X14, 14ll * NKe,
      nullptr, nullptr, nullptr, nullptr, nullptr, 0, xcat, 512,
      stA + 2048, stA + 3072, bn1_g, bn1_b, 2, nullptr, nullptr, 14, 64, NKe, Kk);

  // -------- PaRI conv stages --------
  const int cins[3] = {64, 64, 128}, couts[3] = {64, 128, 256};
  const int chin[3] = {0, 64, 128}, chout[3] = {64, 128, 256};
  for (int li = 0; li < 3; ++li) {
    int base = 7 + 12 * li;
    const float* basisW = (const float*)d_in[base + 0];
    const float* dkw1 = (const float*)d_in[base + 1];
    const float* dkb1 = (const float*)d_in[base + 2];
    const float* dkg  = (const float*)d_in[base + 3];
    const float* dkbb = (const float*)d_in[base + 4];
    const float* dkw2 = (const float*)d_in[base + 5];
    const float* dkb2 = (const float*)d_in[base + 6];
    const float* actg = (const float*)d_in[base + 7];
    const float* actb = (const float*)d_in[base + 8];
    const float* ecw  = (const float*)d_in[base + 9];
    const float* ecg  = (const float*)d_in[base + 10];
    const float* ecb  = (const float*)d_in[base + 11];
    const int cin = cins[li], cout = couts[li];
    const float* xf = xcat + (size_t)chin[li] * Nn;

    knn(xf, 512ll * Nn, cin);
    k_appf<<<dim3((EDGE + 255) / 256), 256, 0, stream>>>(pos, l0, l1, idxb, Ab);

    // h = dk_w1 @ A + b1  (store pre-BN + stats)
    fill(stA, 0.f, 2048);
    k_gemm<GM_STORE | GM_STATS, false><<<dim3(NKe / 128, (cin / 2) / 16, Bb), 256, 0, stream>>>(
        dkw1, 8, dkb1, Ab, 8ll * NKe,
        nullptr, nullptr, nullptr, nullptr, nullptr, 0, hbuf, cin / 2,
        nullptr, nullptr, nullptr, nullptr, 0, stA, stA + 1024, 8, cin / 2, NKe, Kk);
    k_finstat<<<dim3(4), 256, 0, stream>>>(stA, stA + 1024, stA + 2048, stA + 3072,
                                           (float)(Bb * NKe), cin / 2);
    // dk = dk_w2 @ relu(BN(h)) + b2  (input-BN fused, TDM staging)
    k_gemm<GM_STORE, true><<<dim3(NKe / 128, cin / 16, Bb), 256, 0, stream>>>(
        dkw2, cin / 2, dkb2, hbuf, (long long)(cin / 2) * NKe,
        stA + 2048, stA + 3072, dkg, dkbb, nullptr, 0, dkbuf, cin,
        nullptr, nullptr, nullptr, nullptr, 0, nullptr, nullptr, cin / 2, cin, NKe, Kk);
    // basis = basisW @ xf
    k_gemm<GM_STORE, false><<<dim3(Nn / 128, cin / 16, Bb), 256, 0, stream>>>(
        basisW, cin, nullptr, xf, 512ll * Nn,
        nullptr, nullptr, nullptr, nullptr, nullptr, 0, basb, cin,
        nullptr, nullptr, nullptr, nullptr, 0, nullptr, nullptr, cin, cin, Nn, 1);
    // feat_pre = dk * gather(basis) (+ stats for act-BN)
    fill(stB, 0.f, 2048);
    k_gathermul<<<dim3((unsigned)(((size_t)Bb * cin * NKe + 255) / 256)), 256, 0, stream>>>(
        dkbuf, basb, idxb, cin, stB, stB + 1024);
    k_finstat<<<dim3(4), 256, 0, stream>>>(stB, stB + 1024, stB + 2048, stB + 3072,
                                           (float)(Bb * NKe), cin);
    // e = W1@feat + (W2-W1)@xf : broadcast term q over k
    k_makewq<<<dim3((cout * cin + 255) / 256), 256, 0, stream>>>(ecw, cin, cout, wqb);
    k_gemm<GM_STORE, false><<<dim3(Nn / 128, cout / 16, Bb), 256, 0, stream>>>(
        wqb, cin, nullptr, xf, 512ll * Nn,
        nullptr, nullptr, nullptr, nullptr, nullptr, 0, qbuf, cout,
        nullptr, nullptr, nullptr, nullptr, 0, nullptr, nullptr, cin, cout, Nn, 1);
    // pass 1: stats of e (recompute strategy avoids 335MB intermediate)
    fill(stC, 0.f, 2048);
    k_gemm<GM_STATS, true><<<dim3(NKe / 128, cout / 16, Bb), 256, 0, stream>>>(
        ecw, 2 * cin, nullptr, dkbuf, (long long)cin * NKe,
        stB + 2048, stB + 3072, actg, actb, qbuf, cout, nullptr, 0,
        nullptr, nullptr, nullptr, nullptr, 0, stC, stC + 1024, cin, cout, NKe, Kk);
    k_finstat<<<dim3(4), 256, 0, stream>>>(stC, stC + 1024, stC + 2048, stC + 3072,
                                           (float)(Bb * NKe), cout);
    // pass 2: recompute e, BN + lrelu + max over k -> xcat slice
    k_gemm<GM_MAXK, true><<<dim3(NKe / 128, cout / 16, Bb), 256, 0, stream>>>(
        ecw, 2 * cin, nullptr, dkbuf, (long long)cin * NKe,
        stB + 2048, stB + 3072, actg, actb, qbuf, cout,
        xcat + (size_t)chout[li] * Nn, 512,
        stC + 2048, stC + 3072, ecg, ecb, 2, nullptr, nullptr, cin, cout, NKe, Kk);
  }

  // -------- conv5 + pooling + FC head --------
  fill(stP, 0.f, 2048);
  k_gemm<GM_STORE | GM_STATS, true><<<dim3(Nn / 128, 64, Bb), 256, 0, stream>>>(
      conv5_w, 512, nullptr, xcat, 512ll * Nn,
      nullptr, nullptr, nullptr, nullptr, nullptr, 0, c5b, 1024,
      nullptr, nullptr, nullptr, nullptr, 0, stP, stP + 1024, 512, 1024, Nn, 1);
  k_finstat<<<dim3(4), 256, 0, stream>>>(stP, stP + 1024, stP + 2048, stP + 3072,
                                         (float)(Bb * Nn), 1024);
  k_pool<<<dim3(1024, Bb), 256, 0, stream>>>(c5b, stP + 2048, stP + 3072,
                                             bn5_g, bn5_b, gvec);
  k_fc<<<dim3(1), 256, 0, stream>>>(gvec, l1_w, bn6_g, bn6_b, l2_w, l2_b,
                                    bn7_g, bn7_b, l3_w, l3_b, (float*)d_out);
}
